// MultiHeadContinuousCritic_45724221833342
// MI455X (gfx1250) — matrix operand order repaired and checked by reading, not compiled, not run on hardware
//
#include <hip/hip_runtime.h>

#define NB      65536
#define NT      3
#define NH      256
#define NIN     264
#define K1PAD   288
#define MTILE   128
#define NTHREADS 512
#define WSET_U16 204800   // 288*256 + 256*256 + 256*256 halfwords per (net,task)

typedef unsigned short u16;
typedef unsigned int   u32;
typedef __attribute__((ext_vector_type(16))) __bf16 bf16x16;
typedef __attribute__((ext_vector_type(8)))  float  f32x8;

union FragBF { uint4 u[2]; bf16x16 v; };

__device__ __forceinline__ u16 f2bf(float f) {
  u32 u = __float_as_uint(f);
  u += 0x7FFFu + ((u >> 16) & 1u);   // round-to-nearest-even
  return (u16)(u >> 16);
}
__device__ __forceinline__ float bf2f(u16 h) {
  return __uint_as_float(((u32)h) << 16);
}

__global__ void zero_counts_kernel(int* counts) {
  if (threadIdx.x < 4) counts[threadIdx.x] = 0;
}

// Hard top-1 routing on obs[:, -3:]; first-max tie-break matches jnp.argmax.
__global__ void route_kernel(const float* __restrict__ obs,
                             int* __restrict__ counts, int* __restrict__ perm) {
  int b = blockIdx.x * blockDim.x + threadIdx.x;
  if (b >= NB) return;
  const float* o = obs + (size_t)b * 256 + (256 - NT);
  float m = o[0]; int t = 0;
  if (o[1] > m) { m = o[1]; t = 1; }
  if (o[2] > m) { m = o[2]; t = 2; }
  int pos = atomicAdd(&counts[t], 1);
  perm[t * NB + pos] = b;
}

// f32 [K][N] -> bf16 [n][Kpad] (transposed, zero-padded K) per (net,task,layer).
__global__ void prep_weights_kernel(const float* __restrict__ q1W1, const float* __restrict__ q1W2,
                                    const float* __restrict__ q1W3,
                                    const float* __restrict__ q2W1, const float* __restrict__ q2W2,
                                    const float* __restrict__ q2W3,
                                    u16* __restrict__ wsW) {
  int e = blockIdx.x * blockDim.x + threadIdx.x;
  if (e >= 6 * WSET_U16) return;
  int set = e / WSET_U16;
  int rem = e - set * WSET_U16;
  int q = set / 3, t = set - q * 3;
  const float* W1 = q ? q2W1 : q1W1;
  const float* W2 = q ? q2W2 : q1W2;
  const float* W3 = q ? q2W3 : q1W3;
  float v;
  if (rem < K1PAD * 256) {
    int n = rem / K1PAD, k = rem - n * K1PAD;
    v = (k < NIN) ? W1[((size_t)t * NIN + k) * NH + n] : 0.0f;
  } else if (rem < K1PAD * 256 + 65536) {
    int r2 = rem - K1PAD * 256;
    int n = r2 >> 8, k = r2 & 255;
    v = W2[((size_t)t * NH + k) * NH + n];
  } else {
    int r3 = rem - K1PAD * 256 - 65536;
    int n = r3 >> 8, k = r3 & 255;
    v = W3[((size_t)t * NH + k) * NH + n];
  }
  wsW[e] = f2bf(v);
}

// Copy one layer's bf16 weights [256][Kpad] -> LDS with +8 halfword row padding.
__device__ __forceinline__ void load_w_lds(const u16* __restrict__ g, u16* __restrict__ sW,
                                           int Kpad, int tid) {
  const int cpr = Kpad >> 3;          // 16B chunks per row
  const int total = 256 * cpr;
  const int SW = Kpad + 8;
  for (int c = tid; c < total; c += NTHREADS) {
    int n = c / cpr, kc = c - n * cpr;
    *(uint4*)(sW + n * SW + (kc << 3)) = *(const uint4*)(g + n * Kpad + (kc << 3));
  }
}

// One 128x256 GEMM layer: per-wave 16(M) x 128(N) stripe, bias-init accumulators,
// v_wmma_f32_16x16x32_bf16 with ISA-layout fragments, relu + bf16 store.
template <int KPAD>
__device__ __forceinline__ void gemm_tile(const u16* __restrict__ aIn, int saIn,
                                          const u16* __restrict__ sW,
                                          const float* __restrict__ bias,
                                          u16* __restrict__ aOut, int saOut,
                                          int msub, int nhalf, int lane) {
  const int SW = KPAD + 8;
  const int lmod = lane & 15;
  const int lhi  = lane >> 4;
  f32x8 acc[8];
#pragma unroll
  for (int j = 0; j < 8; ++j) {
    float bv = bias[nhalf * 128 + j * 16 + lmod];
#pragma unroll
    for (int r = 0; r < 8; ++r) acc[j][r] = bv;
  }
  const u16* aRow = aIn + (msub * 16 + lmod) * saIn;   // A: M = lane%16
  for (int k0 = 0; k0 < KPAD; k0 += 32) {
    FragBF a;                                          // A elems: K=(e&7)+8*lhi+16*(e>>3)
    a.u[0] = *(const uint4*)(aRow + k0 + lhi * 8);
    a.u[1] = *(const uint4*)(aRow + k0 + 16 + lhi * 8);
#pragma unroll
    for (int j = 0; j < 8; ++j) {
      FragBF b;                                        // B elems: K=(e&7)+8*lhi+16*(e>>3), N=lane%16
      const u16* bRow = sW + (nhalf * 128 + j * 16 + lmod) * SW + k0 + lhi * 8;
      b.u[0] = *(const uint4*)bRow;
      b.u[1] = *(const uint4*)(bRow + 16);
      acc[j] = __builtin_amdgcn_wmma_f32_16x16x32_bf16(false, a.v, false, b.v,
                                                       (short)0, acc[j], false, false);
    }
  }
#pragma unroll
  for (int j = 0; j < 8; ++j) {
    int n = nhalf * 128 + j * 16 + lmod;
#pragma unroll
    for (int r = 0; r < 8; ++r) {                      // C: M = r + 8*lhi
      int m = msub * 16 + lhi * 8 + r;
      float v = acc[j][r];
      v = v > 0.0f ? v : 0.0f;
      aOut[m * saOut + n] = f2bf(v);
    }
  }
}

__global__ __launch_bounds__(NTHREADS, 1)
void mlp_kernel(const float* __restrict__ obs, const float* __restrict__ actions,
                const u16* __restrict__ wsW, const int* __restrict__ counts,
                const int* __restrict__ perm,
                const float* __restrict__ q1b1, const float* __restrict__ q1b2,
                const float* __restrict__ q1b3, const float* __restrict__ q1W4,
                const float* __restrict__ q1b4,
                const float* __restrict__ q2b1, const float* __restrict__ q2b2,
                const float* __restrict__ q2b3, const float* __restrict__ q2W4,
                const float* __restrict__ q2b4,
                float* __restrict__ out) {
  const int q   = blockIdx.z;
  const int t   = blockIdx.y;
  const int off = blockIdx.x * MTILE;
  const int cnt = counts[t];
  if (off >= cnt) return;                 // uniform early-exit for empty tiles
  int rows = cnt - off; if (rows > MTILE) rows = MTILE;

  extern __shared__ char smem[];
  u16* sW   = (u16*)smem;                                   // 256 x 296 bf16 = 151552 B
  u16* bufA = (u16*)(smem + 256 * 296 * 2);                 // 128 x 296 bf16 =  75776 B
  u16* bufB = (u16*)(smem + 256 * 296 * 2 + 128 * 296 * 2); // 128 x 264 bf16 =  67584 B

  const int tid   = threadIdx.x;
  const int lane  = tid & 31;
  const int wave  = tid >> 5;
  const int msub  = wave & 7;
  const int nhalf = wave >> 3;

  const int* permT = perm + t * NB + off;

  // Gather routed rows of x = [obs | actions], f32 -> bf16, K zero-padded to 288.
  {
    int r = tid & 127;
    int p = tid >> 7;                 // 4 threads per row, 72 cols each (4*72 = 288)
    int gi = (r < rows) ? permT[r] : -1;
    u16* dst = bufA + r * 296;
    int ks = p * 72;
    for (int k = ks; k < ks + 72; ++k) {
      float v = 0.0f;
      if (gi >= 0) {
        if (k < 256)      v = obs[(size_t)gi * 256 + k];
        else if (k < NIN) v = actions[(size_t)gi * 8 + (k - 256)];
      }
      dst[k] = f2bf(v);
    }
  }

  const u16* wset = wsW + (size_t)(q * 3 + t) * WSET_U16;
  const float* b1 = (q ? q2b1 : q1b1) + t * NH;
  const float* b2 = (q ? q2b2 : q1b2) + t * NH;
  const float* b3 = (q ? q2b3 : q1b3) + t * NH;
  const float* w4 = (q ? q2W4 : q1W4) + t * NH;
  const float  b4 = (q ? q2b4 : q1b4)[t];

  load_w_lds(wset, sW, K1PAD, tid);
  __syncthreads();
  gemm_tile<K1PAD>(bufA, 296, sW, b1, bufB, 264, msub, nhalf, lane);
  __syncthreads();
  load_w_lds(wset + K1PAD * 256, sW, 256, tid);
  __syncthreads();
  gemm_tile<256>(bufB, 264, sW, b2, bufA, 296, msub, nhalf, lane);
  __syncthreads();
  load_w_lds(wset + K1PAD * 256 + 65536, sW, 256, tid);
  __syncthreads();
  gemm_tile<256>(bufA, 296, sW, b3, bufB, 264, msub, nhalf, lane);
  __syncthreads();

  // Layer 4: H -> 1 dot product, 4 partial sums per row, then scatter to output.
  {
    float* red = (float*)sW;          // sW free after last barrier
    int r = tid >> 2, p = tid & 3;
    const u16* hrow = bufB + r * 264 + p * 64;
    const float* wv = w4 + p * 64;
    float s = 0.0f;
#pragma unroll 8
    for (int j = 0; j < 64; ++j) s += bf2f(hrow[j]) * wv[j];
    red[(r << 2) + p] = s;
    __syncthreads();
    if (p == 0 && r < rows) {
      int gi = permT[r];
      out[(size_t)q * NB + gi] =
          red[r << 2] + red[(r << 2) + 1] + red[(r << 2) + 2] + red[(r << 2) + 3] + b4;
    }
  }
}

extern "C" void kernel_launch(void* const* d_in, const int* in_sizes, int n_in,
                              void* d_out, int out_size, void* d_ws, size_t ws_size,
                              hipStream_t stream) {
  (void)in_sizes; (void)n_in; (void)out_size; (void)ws_size;
  const float* obs     = (const float*)d_in[0];
  const float* actions = (const float*)d_in[1];
  const float* q1W1 = (const float*)d_in[2];
  const float* q1b1 = (const float*)d_in[3];
  const float* q1W2 = (const float*)d_in[4];
  const float* q1b2 = (const float*)d_in[5];
  const float* q1W3 = (const float*)d_in[6];
  const float* q1b3 = (const float*)d_in[7];
  const float* q1W4 = (const float*)d_in[8];
  const float* q1b4 = (const float*)d_in[9];
  const float* q2W1 = (const float*)d_in[10];
  const float* q2b1 = (const float*)d_in[11];
  const float* q2W2 = (const float*)d_in[12];
  const float* q2b2 = (const float*)d_in[13];
  const float* q2W3 = (const float*)d_in[14];
  const float* q2b3 = (const float*)d_in[15];
  const float* q2W4 = (const float*)d_in[16];
  const float* q2b4 = (const float*)d_in[17];

  int* counts = (int*)d_ws;
  int* perm   = (int*)((char*)d_ws + 16);
  u16* wsW    = (u16*)((char*)d_ws + 16 + (size_t)NT * NB * 4);

  zero_counts_kernel<<<1, 32, 0, stream>>>(counts);
  route_kernel<<<NB / 256, 256, 0, stream>>>(obs, counts, perm);
  prep_weights_kernel<<<(6 * WSET_U16 + 255) / 256, 256, 0, stream>>>(
      q1W1, q1W2, q1W3, q2W1, q2W2, q2W3, wsW);

  dim3 grid(NB / MTILE, NT, 2);     // worst-case tiles per task; empty blocks exit
  mlp_kernel<<<grid, NTHREADS, 294912, stream>>>(
      obs, actions, wsW, counts, perm,
      q1b1, q1b2, q1b3, q1W4, q1b4,
      q2b1, q2b2, q2b3, q2W4, q2b4,
      (float*)d_out);
}
